// SNN_FC_66271345377646
// MI455X (gfx1250) — matrix operand order repaired and checked by reading, not compiled, not run on hardware
//
#include <hip/hip_runtime.h>

typedef __attribute__((ext_vector_type(16))) _Float16 v16h;
typedef __attribute__((ext_vector_type(8)))  _Float16 v8h;
typedef __attribute__((ext_vector_type(8)))  float    v8f;
typedef __attribute__((ext_vector_type(4)))  float    v4f;

#define T_STEPS 8
#define B_TOT   16384
#define D_IN    784
#define H_DIM   100
#define C_DIM   10
#define NT      7      // 112 padded H / 16
#define MT      2      // M-tiles per wave (32 batch rows) -> B-frag reuse x2
#define KC1     25     // 800 padded D_IN / 32
#define KP1     808    // LDS row stride (halfs) for W1
#define OS      136    // scratch row stride (halfs)
#define W2S     136    // W2 LDS row stride (halfs)
#define NWAVE   4      // waves per block
#define ROWS_PER_BLOCK 128   // NWAVE * MT * 16

__device__ __forceinline__ void spike8(const float* __restrict__ xp,
                                       const float* __restrict__ rp,
                                       v16h& a, int o) {
  v4f x0 = *(const v4f*)xp;       v4f x1 = *(const v4f*)(xp + 4);
  v4f r0 = *(const v4f*)rp;       v4f r1 = *(const v4f*)(rp + 4);
#pragma unroll
  for (int j = 0; j < 4; ++j) {
    float x = x0[j];
    float s = (2.0f * r0[j] <= __builtin_fabsf(x))
                ? ((x > 0.f) ? 1.f : ((x < 0.f) ? -1.f : 0.f)) : 0.f;
    a[o + j] = (_Float16)s;
    x = x1[j];
    s = (2.0f * r1[j] <= __builtin_fabsf(x))
                ? ((x > 0.f) ? 1.f : ((x < 0.f) ? -1.f : 0.f)) : 0.f;
    a[o + 4 + j] = (_Float16)s;
  }
}

__device__ __forceinline__ v16h lds_bfrag(const _Float16* p) {
  v8h lo = *(const v8h*)p;
  v8h hi = *(const v8h*)(p + 16);
  v16h b;
#pragma unroll
  for (int i = 0; i < 8; ++i) { b[i] = lo[i]; b[8 + i] = hi[i]; }
  return b;
}

__global__ __launch_bounds__(32 * NWAVE) void snn_fused_kernel(
    const float* __restrict__ inp, const float* __restrict__ rnd,
    const float* __restrict__ W1,  const float* __restrict__ b1,
    const float* __restrict__ W2,  const float* __restrict__ b2,
    float* __restrict__ out)
{
  __shared__ _Float16 sW1[112 * KP1];          // [h][k] f16, zero-padded
  __shared__ _Float16 sW2[16 * W2S];           // [c][h] f16, zero-padded
  __shared__ _Float16 sOut[NWAVE][32 * OS];    // per-wave spike-out scratch (32 rows)
  __shared__ float    sB1[112];

  const int tid  = threadIdx.x;
  const int wave = tid >> 5;
  const int lane = tid & 31;
  const int nthr = 32 * NWAVE;

  // ---- stage weights into LDS (f32 -> f16, zero pad) ----
  for (int i = tid; i < 112 * KP1; i += nthr) {
    int h = i / KP1, k = i - h * KP1;
    float v = (h < H_DIM && k < D_IN) ? W1[h * D_IN + k] : 0.0f;
    sW1[i] = (_Float16)v;
  }
  for (int i = tid; i < 16 * W2S; i += nthr) {
    int c = i / W2S, k = i - c * W2S;
    float v = (c < C_DIM && k < H_DIM) ? W2[c * H_DIM + k] : 0.0f;
    sW2[i] = (_Float16)v;
  }
  if (tid < 112) sB1[tid] = (tid < H_DIM) ? b1[tid] : 0.0f;
  for (int i = tid; i < NWAVE * 32 * OS; i += nthr) (&sOut[0][0])[i] = (_Float16)0.0f;
  __syncthreads();

  const int m    = lane & 15;                  // tile row / column selector
  const int hs   = lane >> 4;                  // half-wave select
  const int row0 = blockIdx.x * ROWS_PER_BLOCK + wave * (MT * 16);

  const float* inpRowT[MT];
#pragma unroll
  for (int mt = 0; mt < MT; ++mt)
    inpRowT[mt] = inp + (size_t)(row0 + mt * 16 + m) * D_IN;

  v8f m1[MT][NT];
#pragma unroll
  for (int mt = 0; mt < MT; ++mt)
#pragma unroll
    for (int nt = 0; nt < NT; ++nt) m1[mt][nt] = (v8f)0.0f;
  v8f macc[MT];
#pragma unroll
  for (int mt = 0; mt < MT; ++mt) macc[mt] = (v8f)0.0f;

  _Float16* myScr = &sOut[wave][0];

  for (int t = 0; t < T_STEPS; ++t) {
    const float* randRowT[MT];
#pragma unroll
    for (int mt = 0; mt < MT; ++mt)
      randRowT[mt] = rnd + ((size_t)t * B_TOT + row0 + mt * 16 + m) * D_IN;

    // leak + bias: seed accumulators with 0.95*mem1 + b1
#pragma unroll
    for (int nt = 0; nt < NT; ++nt) {
      float bv = sB1[nt * 16 + m];
#pragma unroll
      for (int mt = 0; mt < MT; ++mt)
#pragma unroll
        for (int r = 0; r < 8; ++r) m1[mt][nt][r] = 0.95f * m1[mt][nt][r] + bv;
    }

    // ---- GEMM1: mem1 += spike[32x800] @ W1t[800x112] ----
    for (int kc = 0; kc < KC1; ++kc) {
      const int kb0 = kc * 32 + hs * 8;
      const int kb1 = kb0 + 16;

      v16h a[MT];
#pragma unroll
      for (int mt = 0; mt < MT; ++mt) {
        if (kc + 4 < KC1)
          __builtin_prefetch(randRowT[mt] + (kc + 4) * 32, 0, 1); // global_prefetch_b8
        spike8(inpRowT[mt] + kb0, randRowT[mt] + kb0, a[mt], 0);
        if (kb1 < D_IN) {
          spike8(inpRowT[mt] + kb1, randRowT[mt] + kb1, a[mt], 8);
        } else {
#pragma unroll
          for (int j = 0; j < 8; ++j) a[mt][8 + j] = (_Float16)0.0f;
        }
      }

#pragma unroll
      for (int nt = 0; nt < NT; ++nt) {
        const _Float16* bp = &sW1[(size_t)(nt * 16 + m) * KP1 + kc * 32 + hs * 8];
        v16h b = lds_bfrag(bp);     // one B-frag feeds MT WMMAs
#pragma unroll
        for (int mt = 0; mt < MT; ++mt)
          m1[mt][nt] = __builtin_amdgcn_wmma_f32_16x16x32_f16(
              false, a[mt], false, b, (short)0, m1[mt][nt], false, false);
      }
    }

    // ---- threshold + re-layout D->A via LDS scratch ----
#pragma unroll
    for (int mt = 0; mt < MT; ++mt)
#pragma unroll
      for (int nt = 0; nt < NT; ++nt)
#pragma unroll
        for (int r = 0; r < 8; ++r) {
          float v = m1[mt][nt][r];
          float o = (v > 1.0f) ? 1.0f : 0.0f;
          m1[mt][nt][r] = v - o;
          myScr[(size_t)(mt * 16 + r + hs * 8) * OS + nt * 16 + m] = (_Float16)o;
        }

    // ---- GEMM2: mem2 += out[32x112] @ W2t[112x16] (C padded to 16) ----
#pragma unroll
    for (int kc2 = 0; kc2 < 4; ++kc2) {
      v16h bw = lds_bfrag(&sW2[(size_t)m * W2S + kc2 * 32 + hs * 8]);
#pragma unroll
      for (int mt = 0; mt < MT; ++mt) {
        v16h a2;
        const _Float16* p0 = &myScr[(size_t)(mt * 16 + m) * OS + kc2 * 32 + hs * 8];
        v8h g0 = *(const v8h*)p0;
        v8h g1 = *(const v8h*)(p0 + 16);
#pragma unroll
        for (int i = 0; i < 8; ++i) { a2[i] = g0[i]; a2[8 + i] = g1[i]; }
        macc[mt] = __builtin_amdgcn_wmma_f32_16x16x32_f16(
            false, a2, false, bw, (short)0, macc[mt], false, false);
      }
    }
  }

  // ---- epilogue: out = macc/T + b2 ----
  if (m < C_DIM) {
    float bv = b2[m];
#pragma unroll
    for (int mt = 0; mt < MT; ++mt)
#pragma unroll
      for (int r = 0; r < 8; ++r) {
        int row = row0 + mt * 16 + r + hs * 8;
        out[(size_t)row * C_DIM + m] = macc[mt][r] * (1.0f / T_STEPS) + bv;
      }
  }
}

extern "C" void kernel_launch(void* const* d_in, const int* in_sizes, int n_in,
                              void* d_out, int out_size, void* d_ws, size_t ws_size,
                              hipStream_t stream) {
  const float* inp = (const float*)d_in[0];
  const float* rnd = (const float*)d_in[1];
  const float* W1  = (const float*)d_in[2];
  const float* b1  = (const float*)d_in[3];
  const float* W2  = (const float*)d_in[4];
  const float* b2  = (const float*)d_in[5];
  float* out = (float*)d_out;
  snn_fused_kernel<<<B_TOT / ROWS_PER_BLOCK, 32 * NWAVE, 0, stream>>>(
      inp, rnd, W1, b1, W2, b2, out);
}
